// UTF8CodeBook_11776800326326
// MI455X (gfx1250) — matrix operand order repaired and checked by reading, not compiled, not run on hardware
//
#include <hip/hip_runtime.h>
#include <stdint.h>

// Problem constants (reference: N=4096, V=65536, D=128, k=1)
#define DD 128
#define TILE_M 16
#define WAVES_PER_BLOCK 8
#define COLS_PER_ITER (WAVES_PER_BLOCK * 16)   // 128 codebook cols per block-iteration
#define NSLICE 8                               // V split for extra grid parallelism

typedef __attribute__((ext_vector_type(16))) __bf16 v16bf;
typedef __attribute__((ext_vector_type(8)))  float  v8f;

union BfPack {          // 8 dwords = 16 bf16 = one WMMA 16-bit operand per lane
    uint32_t u[8];
    uint4    q[2];
    v16bf    v;
};

__device__ __forceinline__ uint16_t bf_rne(float f) {
    uint32_t u = __float_as_uint(f);
    return (uint16_t)((u + 0x7FFFu + ((u >> 16) & 1u)) >> 16);
}
__device__ __forceinline__ float bf2f(uint16_t h) {
    return __uint_as_float(((uint32_t)h) << 16);
}
__device__ __forceinline__ uint32_t pack2(uint16_t a, uint16_t b) {
    return (uint32_t)a | ((uint32_t)b << 16);
}

// One wave per row of 128 floats: split into bf16 hi/lo (packed 2-per-dword),
// optional squared-norm via wave32 shuffle reduction.
__global__ void split_bf16_kernel(const float* __restrict__ src,
                                  uint32_t* __restrict__ hi,
                                  uint32_t* __restrict__ lo,
                                  float* __restrict__ sq, int rows)
{
    int wave = (int)((blockIdx.x * blockDim.x + threadIdx.x) >> 5);
    int lane = (int)(threadIdx.x & 31);
    if (wave >= rows) return;
    const float4 f = *(const float4*)(src + (size_t)wave * DD + lane * 4);

    uint16_t h0 = bf_rne(f.x), h1 = bf_rne(f.y), h2 = bf_rne(f.z), h3 = bf_rne(f.w);
    uint16_t l0 = bf_rne(f.x - bf2f(h0));
    uint16_t l1 = bf_rne(f.y - bf2f(h1));
    uint16_t l2 = bf_rne(f.z - bf2f(h2));
    uint16_t l3 = bf_rne(f.w - bf2f(h3));

    size_t base = (size_t)wave * (DD / 2) + (size_t)lane * 2;
    hi[base + 0] = pack2(h0, h1);
    hi[base + 1] = pack2(h2, h3);
    lo[base + 0] = pack2(l0, l1);
    lo[base + 1] = pack2(l2, l3);

    if (sq) {
        float p = f.x * f.x + f.y * f.y + f.z * f.z + f.w * f.w;
        #pragma unroll
        for (int m = 16; m >= 1; m >>= 1) p += __shfl_xor(p, m, 32);
        if (lane == 0) sq[wave] = p;
    }
}

// Main fused GEMM + argmin kernel.
// Block: 256 threads (8 waves). blockIdx.x -> 16-row M tile, blockIdx.y -> V slice.
// Each wave: 16 columns/iteration, K=128 as 4 chunks of 32, 3 bf16 passes
// (hi*hi + hi*lo + lo*hi) -> 12 v_wmma_f32_16x16x32_bf16 per iteration.
__global__ void __launch_bounds__(256)
knn_wmma_kernel(const uint32_t* __restrict__ xhi, const uint32_t* __restrict__ xlo,
                const uint32_t* __restrict__ chi, const uint32_t* __restrict__ clo,
                const float* __restrict__ csq,
                float* __restrict__ part_s, int* __restrict__ part_i,
                int sliceCols)
{
    const int m0   = (int)blockIdx.x * TILE_M;
    const int slc  = (int)blockIdx.y;
    const int tid  = (int)threadIdx.x;
    const int wave = tid >> 5;
    const int lane = tid & 31;
    const int ln   = lane & 15;   // column (B/N) lane within half-wave
    const int hs   = lane >> 4;   // half-wave select (K / row-group split)

    __shared__ float s_best[WAVES_PER_BLOCK * TILE_M];
    __shared__ int   s_idx [WAVES_PER_BLOCK * TILE_M];

    // ---- A operands (x tile rows, hi+lo for all K=128) resident in VGPRs ----
    // ISA 16-bit A layout: lane holds row M=ln; VGPR j<4 -> K dwords hs*4+j,
    // VGPR j>=4 -> K dwords 8+hs*4+(j-4): two contiguous uint4 loads per chunk.
    BfPack ahi[4], alo[4];
    {
        const uint32_t* xh = xhi + (size_t)(m0 + ln) * (DD / 2);
        const uint32_t* xl = xlo + (size_t)(m0 + ln) * (DD / 2);
        #pragma unroll
        for (int kb = 0; kb < 4; ++kb) {
            const int b = kb * 16 + hs * 4;
            ahi[kb].q[0] = *(const uint4*)(xh + b);
            ahi[kb].q[1] = *(const uint4*)(xh + b + 8);
            alo[kb].q[0] = *(const uint4*)(xl + b);
            alo[kb].q[1] = *(const uint4*)(xl + b + 8);
        }
    }

    float bs[8];
    int   bi[8];
    #pragma unroll
    for (int r = 0; r < 8; ++r) { bs[r] = 3.4e38f; bi[r] = 0; }

    const int vslice0 = slc * sliceCols;
    const int iters   = sliceCols / COLS_PER_ITER;

    // Pointer-increment addressing: fixed stride per iteration, no per-iter
    // address recomputation from the loop counter.
    int v = vslice0 + wave * 16 + ln;
    const uint32_t* bhp = chi + (size_t)v * (DD / 2);
    const uint32_t* blp = clo + (size_t)v * (DD / 2);
    const float*    csp = csq + v;
    const size_t bstride = (size_t)COLS_PER_ITER * (DD / 2);

    for (int it = 0; it < iters; ++it) {
        // ISA 16-bit B layout: lane holds column N=ln; VGPR j -> K dword
        // kb*16 + hs*4 + j (j<4), kb*16 + 8 + hs*4 + (j-4) (j>=4): two uint4 loads per chunk.
        BfPack bhi[4], blo_[4];
        #pragma unroll
        for (int kb = 0; kb < 4; ++kb) {
            const int b = kb * 16 + hs * 4;
            bhi[kb].q[0]  = *(const uint4*)(bhp + b);
            bhi[kb].q[1]  = *(const uint4*)(bhp + b + 8);
            blo_[kb].q[0] = *(const uint4*)(blp + b);
            blo_[kb].q[1] = *(const uint4*)(blp + b + 8);
        }
        const float cs = *csp;
        if (it + 1 < iters) {   // pull next iteration's B tile toward L0/L2
            __builtin_prefetch(bhp + bstride, 0, 0);
            __builtin_prefetch(blp + bstride, 0, 0);
        }

        v8f acc = {};
        #pragma unroll
        for (int kb = 0; kb < 4; ++kb) {
            acc = __builtin_amdgcn_wmma_f32_16x16x32_bf16(
                false, ahi[kb].v, false, bhi[kb].v, (short)0, acc, false, false);
            acc = __builtin_amdgcn_wmma_f32_16x16x32_bf16(
                false, ahi[kb].v, false, blo_[kb].v, (short)0, acc, false, false);
            acc = __builtin_amdgcn_wmma_f32_16x16x32_bf16(
                false, alo[kb].v, false, bhi[kb].v, (short)0, acc, false, false);
        }

        // C layout: lane ln = column, VGPR r = row hs*8+r. Fused score+argmin.
        // Per-lane v is strictly increasing, so on an exact score tie the
        // stored (earlier, smaller) index is already correct: plain '<' works.
        #pragma unroll
        for (int r = 0; r < 8; ++r) {
            const float s = fmaf(-2.0f, acc[r], cs);   // c_sq - 2*cross
            if (s < bs[r]) { bs[r] = s; bi[r] = v; }
        }

        v   += COLS_PER_ITER;
        bhp += bstride;
        blp += bstride;
        csp += COLS_PER_ITER;
    }

    // Reduce across the 16 lanes of each half-wave (same 8 rows, different cols).
    // Indices interleave across lanes here, so keep the tie-aware compare.
    #pragma unroll
    for (int m = 1; m <= 8; m <<= 1) {
        #pragma unroll
        for (int r = 0; r < 8; ++r) {
            const float os = __shfl_xor(bs[r], m, 32);
            const int   oi = __shfl_xor(bi[r], m, 32);
            if (os < bs[r] || (os == bs[r] && oi < bi[r])) { bs[r] = os; bi[r] = oi; }
        }
    }
    if (ln == 0) {
        #pragma unroll
        for (int r = 0; r < 8; ++r) {
            s_best[wave * TILE_M + hs * 8 + r] = bs[r];
            s_idx [wave * TILE_M + hs * 8 + r] = bi[r];
        }
    }
    __syncthreads();

    if (tid < TILE_M) {
        float b = s_best[tid];
        int   i = s_idx[tid];
        #pragma unroll
        for (int w = 1; w < WAVES_PER_BLOCK; ++w) {
            const float os = s_best[w * TILE_M + tid];
            const int   oi = s_idx [w * TILE_M + tid];
            if (os < b || (os == b && oi < i)) { b = os; i = oi; }
        }
        part_s[(size_t)(m0 + tid) * NSLICE + slc] = b;
        part_i[(size_t)(m0 + tid) * NSLICE + slc] = i;
    }
}

__global__ void finalize_kernel(const float* __restrict__ ps,
                                const int* __restrict__ pi,
                                int* __restrict__ out, int rows)
{
    const int row = (int)(blockIdx.x * blockDim.x + threadIdx.x);
    if (row >= rows) return;
    float b = ps[(size_t)row * NSLICE];
    int   i = pi[(size_t)row * NSLICE];
    #pragma unroll
    for (int s = 1; s < NSLICE; ++s) {
        const float os = ps[(size_t)row * NSLICE + s];
        const int   oi = pi[(size_t)row * NSLICE + s];
        if (os < b || (os == b && oi < i)) { b = os; i = oi; }
    }
    out[row] = i;
}

extern "C" void kernel_launch(void* const* d_in, const int* in_sizes, int n_in,
                              void* d_out, int out_size, void* d_ws, size_t ws_size,
                              hipStream_t stream)
{
    (void)n_in; (void)out_size; (void)ws_size;
    const float* x  = (const float*)d_in[0];   // [N, 128] f32
    const float* cb = (const float*)d_in[1];   // [V, 128] f32
    // d_in[2] is k (==1); argmin only, so unused.
    const int N = in_sizes[0] / DD;            // 4096
    const int V = in_sizes[1] / DD;            // 65536

    // Workspace layout (~34.5 MB total): bf16 hi/lo splits + c_sq + partials.
    uint8_t* ws = (uint8_t*)d_ws;
    size_t off = 0;
    auto take = [&](size_t bytes) -> void* {
        void* p = ws + off;
        off = (off + bytes + 255) & ~(size_t)255;
        return p;
    };
    uint32_t* xhi = (uint32_t*)take((size_t)N * DD * 2);
    uint32_t* xlo = (uint32_t*)take((size_t)N * DD * 2);
    uint32_t* chi = (uint32_t*)take((size_t)V * DD * 2);
    uint32_t* clo = (uint32_t*)take((size_t)V * DD * 2);
    float*    csq = (float*)   take((size_t)V * 4);
    float*    pts = (float*)   take((size_t)N * NSLICE * 4);
    int*      pti = (int*)     take((size_t)N * NSLICE * 4);

    // 1) bf16 hi/lo splits (one wave per 128-float row).
    split_bf16_kernel<<<dim3((unsigned)(N / 8)), dim3(256), 0, stream>>>(
        x, xhi, xlo, nullptr, N);
    split_bf16_kernel<<<dim3((unsigned)(V / 8)), dim3(256), 0, stream>>>(
        cb, chi, clo, csq, V);

    // 2) Fused bf16x3 WMMA GEMM + argmin over V slices.
    const int sliceCols = V / NSLICE;          // 8192 -> 64 iterations/block
    knn_wmma_kernel<<<dim3((unsigned)(N / TILE_M), NSLICE), dim3(256), 0, stream>>>(
        xhi, xlo, chi, clo, csq, pts, pti, sliceCols);

    // 3) Fold slice partials into final int32 indices.
    finalize_kernel<<<dim3((unsigned)((N + 255) / 256)), dim3(256), 0, stream>>>(
        pts, pti, (int*)d_out, N);
}
